// RegimeGRU_30176440222439
// MI455X (gfx1250) — matrix operand order repaired and checked by reading, not compiled, not run on hardware
//
#include <hip/hip_runtime.h>
#include <hip/hip_bf16.h>

// ---- problem constants ----
constexpr int kB = 256;   // batch
constexpr int kT = 512;   // time
constexpr int kF = 128;   // input features
constexpr int kH = 128;   // hidden
constexpr int kR = 8;     // regimes
constexpr int kK = 64;    // head width
constexpr int k3H = 384;  // 3*H
constexpr int kTile = 16 * k3H;   // floats per (16-row x 384) gi tile = 6144

typedef __attribute__((ext_vector_type(16))) __bf16 v16bf;
typedef __attribute__((ext_vector_type(8)))  float  v8f;
typedef __attribute__((ext_vector_type(4)))  int    v4i_;

union FragBF {           // 16 bf16 = 32 bytes = one WMMA A/B operand per lane
    v16bf v;
    uint4 u[2];
};

static __device__ __forceinline__ float sigmoidf_(float x) {
    return 1.0f / (1.0f + __expf(-x));
}

// 16-byte async global->LDS copy (gfx1250 GLOBAL_LOAD_ASYNC_TO_LDS_B128, ASYNCcnt),
// with a synchronous fallback so compilation can never break.
static __device__ __forceinline__ void async_ld16(const float* g, float* l) {
#if defined(__gfx1250__) && __has_builtin(__builtin_amdgcn_global_load_async_to_lds_b128)
    // builtin signature (from clang diagnostic): (int4 AS1*, int4 AS3*, imm offset, imm cpol)
    __builtin_amdgcn_global_load_async_to_lds_b128(
        (__attribute__((address_space(1))) v4i_*)(const __attribute__((address_space(1))) void*)g,
        (__attribute__((address_space(3))) v4i_*)(__attribute__((address_space(3))) void*)l,
        0, 0);
#else
    *(uint4*)l = *(const uint4*)g;
#endif
}

static __device__ __forceinline__ void wait_async0() {
#if defined(__gfx1250__)
    asm volatile("s_wait_asynccnt 0x0" ::: "memory");
#endif
}

// ---------------- conversion kernels ----------------
__global__ void cvt_f32_bf16(const float* __restrict__ src, __bf16* __restrict__ dst, int n) {
    int i = blockIdx.x * blockDim.x + threadIdx.x;
    int stride = gridDim.x * blockDim.x;
    for (; i < n; i += stride) dst[i] = (__bf16)src[i];
}

// W1 (R,H,K) f32 -> W1t (R,K,H) bf16 so B-fragments are contiguous rows
__global__ void transpose_w1(const float* __restrict__ W1, __bf16* __restrict__ W1t) {
    int i = blockIdx.x * blockDim.x + threadIdx.x;
    if (i >= kR * kK * kH) return;
    int h = i % kH;
    int k = (i / kH) % kK;
    int r = i / (kH * kK);
    W1t[i] = (__bf16)W1[((size_t)r * kH + h) * kK + k];
}

// ---------------- gi = A(n,128) @ Wih^T(128,384) + bih,  stored (t,b,384) f32 ----------------
// 8192 blocks x 256 threads (8 waves); wave owns 3 N-tiles; K loop = 4 x 32.
__global__ void __launch_bounds__(256) gemm_gi_kernel(const __bf16* __restrict__ A,
                                                      const __bf16* __restrict__ Wih,
                                                      const float*  __restrict__ bih,
                                                      float*        __restrict__ gi) {
    const int lane = threadIdx.x & 31;
    const int wave = threadIdx.x >> 5;
    const int n0 = blockIdx.x * 16;
    const int m = lane & 15;
    const int hi = lane >> 4;
    const int half8 = hi * 8, half16 = hi * 16;

    // A fragments (16x32 bf16, contiguous layout: low lanes K0..15, high lanes K16..31)
    FragBF a[4];
    const __bf16* arow = A + (size_t)(n0 + m) * kF;
#pragma unroll
    for (int kt = 0; kt < 4; kt++) {
        a[kt].u[0] = *(const uint4*)(arow + kt * 32 + half16);
        a[kt].u[1] = *(const uint4*)(arow + kt * 32 + half16 + 8);
    }

#pragma unroll
    for (int j = 0; j < 3; j++) {
        const int nt = wave * 3 + j;                 // N-tile (0..23)
        const __bf16* wrow = Wih + (size_t)(nt * 16 + m) * kF;
        v8f acc = {};
#pragma unroll
        for (int kt = 0; kt < 4; kt++) {
            FragBF b;
            b.u[0] = *(const uint4*)(wrow + kt * 32 + half16);
            b.u[1] = *(const uint4*)(wrow + kt * 32 + half16 + 8);
            acc = __builtin_amdgcn_wmma_f32_16x16x32_bf16(false, a[kt].v, false, b.v,
                                                          (short)0, acc, false, false);
        }
        const int col = nt * 16 + m;
        const float bias = bih[col];
#pragma unroll
        for (int i = 0; i < 8; i++) {
            int n = n0 + i + half8;                  // n = b*T + t
            int t = n & (kT - 1);
            int b = n >> 9;
            // gi (201MB) is write-once/read-once streaming: NT store keeps L2 for weights
            __builtin_nontemporal_store(acc[i] + bias,
                                        &gi[((size_t)t * kB + b) * k3H + col]);
        }
    }
}

// ---------------- sequential GRU recurrence ----------------
// 16 blocks (16 batch rows each) x 256 threads (8 waves, 3 N-tiles/wave).
// Whh B-fragments persist in VGPRs across all T steps; h lives in LDS.
// gi tiles (24KB/step, contiguous) are double-buffered into LDS with async copies
// overlapped with the WMMA chain; drained with s_wait_asynccnt before the barrier.
__global__ void __launch_bounds__(256) gru_rec_kernel(const float*  __restrict__ gi,
                                                      const __bf16* __restrict__ Whh,
                                                      const float*  __restrict__ bhh,
                                                      __bf16*       __restrict__ hout) {
    extern __shared__ __align__(16) char smem[];
    float*  h_f   = (float*) (smem);                 //  8192 B
    __bf16* h_b   = (__bf16*)(smem + 8192);          //  4096 B
    float*  gh    = (float*) (smem + 12288);         // 24576 B
    float*  bhh_s = (float*) (smem + 36864);         //  1536 B
    float*  gibuf = (float*) (smem + 38400);         // 2 x 24576 B  -> total 87552 B

    const int tid = threadIdx.x;
    const int lane = tid & 31, wave = tid >> 5;
    const int m = lane & 15, hi = lane >> 4;
    const int half8 = hi * 8, half16 = hi * 16;
    const int b0 = blockIdx.x * 16;

    for (int i = tid; i < 16 * kH; i += 256) { h_f[i] = 0.0f; h_b[i] = (__bf16)0.0f; }
    for (int i = tid; i < k3H; i += 256) bhh_s[i] = bhh[i];

    // kick off async copy of the t=0 gi tile into buffer 0
    {
        const float* gsrc = gi + (size_t)b0 * k3H;
#pragma unroll
        for (int p = 0; p < 6; p++) {
            const int o = (p * 256 + tid) * 4;       // 16B per thread-chunk
            async_ld16(gsrc + o, gibuf + o);
        }
    }

    // Preload persistent Whh B-fragments: 12 frags = 96 VGPRs/lane
    FragBF bf[3][4];
#pragma unroll
    for (int j = 0; j < 3; j++) {
        const int nt = wave * 3 + j;
        const __bf16* wrow = Whh + (size_t)(nt * 16 + m) * kH;
#pragma unroll
        for (int kt = 0; kt < 4; kt++) {
            bf[j][kt].u[0] = *(const uint4*)(wrow + kt * 32 + half16);
            bf[j][kt].u[1] = *(const uint4*)(wrow + kt * 32 + half16 + 8);
        }
    }
    wait_async0();
    __syncthreads();

    for (int t = 0; t < kT; t++) {
        // overlap: async-prefetch gi tile for step t+1 into the other buffer
        if (t + 1 < kT) {
            const float* gsrc = gi + ((size_t)(t + 1) * kB + b0) * k3H;
            float* ldst = gibuf + ((t + 1) & 1) * kTile;
#pragma unroll
            for (int p = 0; p < 6; p++) {
                const int o = (p * 256 + tid) * 4;
                async_ld16(gsrc + o, ldst + o);
            }
        }

        // rebuild A fragments from current h (bf16 mirror in LDS)
        FragBF a[4];
#pragma unroll
        for (int kt = 0; kt < 4; kt++) {
            const __bf16* hrow = h_b + m * kH + kt * 32 + half16;
            a[kt].u[0] = *(const uint4*)(hrow);
            a[kt].u[1] = *(const uint4*)(hrow + 8);
        }
        // gh = h @ Whh^T : 12 WMMA per wave on the serial critical path
#pragma unroll
        for (int j = 0; j < 3; j++) {
            v8f acc = {};
#pragma unroll
            for (int kt = 0; kt < 4; kt++)
                acc = __builtin_amdgcn_wmma_f32_16x16x32_bf16(false, a[kt].v, false, bf[j][kt].v,
                                                              (short)0, acc, false, false);
            const int col = (wave * 3 + j) * 16 + m;
#pragma unroll
            for (int i = 0; i < 8; i++) gh[(i + half8) * k3H + col] = acc[i];
        }
        __syncthreads();

        // gate math: 2048 elements over 256 threads, gi read from LDS buffer
        const float* gtile = gibuf + (t & 1) * kTile;
#pragma unroll
        for (int p = 0; p < 8; p++) {
            const int e = p * 256 + tid;
            const int row = e >> 7, j = e & 127;
            const int b = b0 + row;
            const int tb = row * k3H + j;
            const float ir = gtile[tb];
            const float iz = gtile[tb + 128];
            const float in = gtile[tb + 256];
            const float hr = gh[tb] + bhh_s[j];
            const float hz = gh[tb + 128] + bhh_s[128 + j];
            const float hn = gh[tb + 256] + bhh_s[256 + j];
            const float r = sigmoidf_(ir + hr);
            const float z = sigmoidf_(iz + hz);
            const float nn = tanhf(in + r * hn);
            const float hnew = (1.0f - z) * nn + z * h_f[row * kH + j];
            h_f[row * kH + j] = hnew;
            h_b[row * kH + j] = (__bf16)hnew;
            hout[((size_t)b * kT + t) * kH + j] = (__bf16)hnew;
        }
        wait_async0();          // next tile resident before anyone re-enters gate phase
        __syncthreads();
    }
}

// ---------------- regime head + lag output ----------------
// 8192 blocks x 1 wave; WMMA only for regimes present in the 16-row tile (uniform branch).
__global__ void __launch_bounds__(32) head_kernel(const __bf16* __restrict__ h2,
                                                  const int*    __restrict__ regime,
                                                  const __bf16* __restrict__ W1t,
                                                  const float*  __restrict__ b1,
                                                  const float*  __restrict__ W2,
                                                  const float*  __restrict__ b2,
                                                  const float*  __restrict__ x,
                                                  const float*  __restrict__ lag_scale,
                                                  const float*  __restrict__ lag_bias,
                                                  float*        __restrict__ out) {
    __shared__ __align__(16) float c[16 * kK];
    __shared__ int regs[16];
    __shared__ unsigned mask;

    const int lane = threadIdx.x;
    const int m = lane & 15, hi = lane >> 4;
    const int half8 = hi * 8, half16 = hi * 16;
    const int n0 = blockIdx.x * 16;

    if (lane == 0) mask = 0;
    __syncthreads();
    if (lane < 16) {
        int r = regime[n0 + lane];
        regs[lane] = r;
        atomicOr(&mask, 1u << r);
    }
    __syncthreads();

    FragBF a[4];
    const __bf16* arow = h2 + (size_t)(n0 + m) * kH;
#pragma unroll
    for (int kt = 0; kt < 4; kt++) {
        a[kt].u[0] = *(const uint4*)(arow + kt * 32 + half16);
        a[kt].u[1] = *(const uint4*)(arow + kt * 32 + half16 + 8);
    }

    const float ls = lag_scale[0], lb = lag_bias[0];
    const unsigned msk = mask;

    for (int r = 0; r < kR; r++) {
        if (!((msk >> r) & 1)) continue;             // uniform per wave -> EXEC stays full
#pragma unroll
        for (int ntk = 0; ntk < 4; ntk++) {          // 64 output cols = 4 N-tiles
            v8f acc = {};
            const int kcol = ntk * 16 + m;
            const __bf16* wrow = W1t + ((size_t)r * kK + kcol) * kH;
#pragma unroll
            for (int kt = 0; kt < 4; kt++) {
                FragBF bfr;
                bfr.u[0] = *(const uint4*)(wrow + kt * 32 + half16);
                bfr.u[1] = *(const uint4*)(wrow + kt * 32 + half16 + 8);
                acc = __builtin_amdgcn_wmma_f32_16x16x32_bf16(false, a[kt].v, false, bfr.v,
                                                              (short)0, acc, false, false);
            }
            const float bias = b1[r * kK + kcol];
#pragma unroll
            for (int i = 0; i < 8; i++) c[(i + half8) * kK + kcol] = acc[i] + bias;
        }
        __syncthreads();
        if (lane < 16 && regs[lane] == r) {
            float s = 0.0f;
#pragma unroll 8
            for (int k = 0; k < kK; k++) {
                float aa = c[lane * kK + k];
                s += (aa * sigmoidf_(aa)) * W2[r * kK + k];   // SiLU . W2
            }
            const int n = n0 + lane;                 // n = b*T + t  == flat (B,T) index
            out[n] = ls * x[(size_t)n * kF] + lb + s + b2[r];
        }
        __syncthreads();
    }
}

// ---------------- host launcher ----------------
extern "C" void kernel_launch(void* const* d_in, const int* in_sizes, int n_in,
                              void* d_out, int out_size, void* d_ws, size_t ws_size,
                              hipStream_t stream) {
    const float* x     = (const float*)d_in[0];
    const int*   reg   = (const int*)  d_in[1];
    const float* lsc   = (const float*)d_in[2];
    const float* lbi   = (const float*)d_in[3];
    const float* Wih0  = (const float*)d_in[4];
    const float* Whh0  = (const float*)d_in[5];
    const float* bih0  = (const float*)d_in[6];
    const float* bhh0  = (const float*)d_in[7];
    const float* Wih1  = (const float*)d_in[8];
    const float* Whh1  = (const float*)d_in[9];
    const float* bih1  = (const float*)d_in[10];
    const float* bhh1  = (const float*)d_in[11];
    const float* W1    = (const float*)d_in[12];
    const float* b1    = (const float*)d_in[13];
    const float* W2    = (const float*)d_in[14];
    const float* b2    = (const float*)d_in[15];
    float* out = (float*)d_out;

    // workspace carve (256B aligned):
    char* ws = (char*)d_ws;
    size_t off = 0;
    auto carve = [&](size_t bytes) -> void* {
        void* p = ws + off;
        off += (bytes + 255) & ~(size_t)255;
        return p;
    };
    __bf16* x_bf   = (__bf16*)carve((size_t)kB * kT * kF * 2);   // 32 MB
    __bf16* wih0_b = (__bf16*)carve((size_t)k3H * kF * 2);
    __bf16* whh0_b = (__bf16*)carve((size_t)k3H * kH * 2);
    __bf16* wih1_b = (__bf16*)carve((size_t)k3H * kH * 2);
    __bf16* whh1_b = (__bf16*)carve((size_t)k3H * kH * 2);
    __bf16* w1t_b  = (__bf16*)carve((size_t)kR * kK * kH * 2);
    float*  gi     = (float*) carve((size_t)kT * kB * k3H * 4);  // 201 MB, reused by both layers
    __bf16* h1_bf  = (__bf16*)carve((size_t)kB * kT * kH * 2);   // 32 MB
    __bf16* h2_bf  = (__bf16*)carve((size_t)kB * kT * kH * 2);   // 32 MB
    (void)ws_size; (void)n_in; (void)in_sizes; (void)out_size;

    const int nX = kB * kT * kF;
    const int nW = k3H * kH;
    cvt_f32_bf16<<<4096, 256, 0, stream>>>(x,    x_bf,   nX);
    cvt_f32_bf16<<<(nW + 255) / 256, 256, 0, stream>>>(Wih0, wih0_b, nW);
    cvt_f32_bf16<<<(nW + 255) / 256, 256, 0, stream>>>(Whh0, whh0_b, nW);
    cvt_f32_bf16<<<(nW + 255) / 256, 256, 0, stream>>>(Wih1, wih1_b, nW);
    cvt_f32_bf16<<<(nW + 255) / 256, 256, 0, stream>>>(Whh1, whh1_b, nW);
    transpose_w1<<<(kR * kK * kH + 255) / 256, 256, 0, stream>>>(W1, w1t_b);

    const int rowTiles = (kB * kT) / 16;       // 8192
    const int recLds = 87552;                  // dynamic LDS: h_f + h_b + gh + bhh + 2 gi tiles

    // layer 0
    gemm_gi_kernel<<<rowTiles, 256, 0, stream>>>(x_bf, wih0_b, bih0, gi);
    gru_rec_kernel<<<kB / 16, 256, recLds, stream>>>(gi, whh0_b, bhh0, h1_bf);
    // layer 1 (gi buffer reused)
    gemm_gi_kernel<<<rowTiles, 256, 0, stream>>>(h1_bf, wih1_b, bih1, gi);
    gru_rec_kernel<<<kB / 16, 256, recLds, stream>>>(gi, whh1_b, bhh1, h2_bf);
    // head + lag
    head_kernel<<<rowTiles, 32, 0, stream>>>(h2_bf, reg, w1t_b, b1, W2, b2,
                                             x, lsc, lbi, out);
}
